// FactorizedSelfAttentionBlock_1743756722765
// MI455X (gfx1250) — matrix operand order repaired and checked by reading, not compiled, not run on hardware
//
#include <hip/hip_runtime.h>
#include <hip/hip_bf16.h>
#include <math.h>

// ---------------------------------------------------------------------------
// Transformer block (LN1 -> QKV -> masked attention -> proj+res -> LN2 ->
// fc1+GELU -> fc2+res) for B=8, A=64, T=128, C=256, H=8, WF=4 on gfx1250.
// GEMMs: v_wmma_f32_16x16x32_bf16 with double-buffered async-to-LDS staging
// (global_load_async_to_lds_b128 / s_wait_asynccnt).
// ---------------------------------------------------------------------------

typedef __attribute__((ext_vector_type(16))) __bf16 v16bf;
typedef __attribute__((ext_vector_type(8)))  __bf16 v8bf;
typedef __attribute__((ext_vector_type(8)))  float  v8f;

#define NTOK   65536   // B*A*T
#define CDIM   256
#define C3     768
#define CFF    1024
#define TSEQ   128
#define NHEAD  8
#define DHEAD  32
#define EPS_LN 1e-5f

__device__ __forceinline__ __bf16 tobf(float f) { return (__bf16)f; }

__device__ __forceinline__ v8f wmma_bf16(v16bf a, v16bf b, v8f c) {
  return __builtin_amdgcn_wmma_f32_16x16x32_bf16(false, a, false, b,
                                                 (short)0, c, false, false);
}

// Async DMA of 16 bytes: global -> LDS, tracked by ASYNCcnt.
__device__ __forceinline__ void async_b128(const __bf16* lds_dst,
                                           const __bf16* gsrc) {
  unsigned lds_off = (unsigned)(uintptr_t)lds_dst;  // low 32b = LDS offset
  unsigned long long ga = (unsigned long long)(uintptr_t)gsrc;
  asm volatile("global_load_async_to_lds_b128 %0, %1, off"
               :: "v"(lds_off), "v"(ga) : "memory");
}

__device__ __forceinline__ void wait_async0() {
  asm volatile("s_wait_asynccnt 0" ::: "memory");
}

// A-matrix fragment (16x32, bf16): lane = row (mod 16); lanes 0-15 hold
// K = koff + 0..15, lanes 16-31 hold K = koff + 16..31.
__device__ __forceinline__ v16bf load_a_frag(const __bf16* base, int stride,
                                             int row0, int koff, int lane) {
  int r = lane & 15, half = lane >> 4;
  const __bf16* p = base + (size_t)(row0 + r) * stride + koff + half * 16;
  v8bf lo = *(const v8bf*)p;          // K = half*16 + 0..7
  v8bf hi = *(const v8bf*)(p + 8);    // K = half*16 + 8..15
  v16bf a;
#pragma unroll
  for (int i = 0; i < 8; ++i) { a[i] = lo[i]; a[8 + i] = hi[i]; }
  return a;
}

// B-matrix fragment (32x16, bf16) from K-contiguous storage Bt[n][k]
// (i.e. the original row-major weight W[n][k]). lane = col n; lanes 0-15
// hold K = 0..15, lanes 16-31 hold K = 16..31 (contiguous 16 elements).
__device__ __forceinline__ v16bf load_bt_frag(const __bf16* base, int stride,
                                              int n0, int lane) {
  int n = lane & 15, half = lane >> 4;
  const __bf16* p = base + (size_t)(n0 + n) * stride + half * 16;
  v8bf lo = *(const v8bf*)p;
  v8bf hi = *(const v8bf*)(p + 8);
  v16bf b;
#pragma unroll
  for (int i = 0; i < 8; ++i) { b[i] = lo[i]; b[8 + i] = hi[i]; }
  return b;
}

// B-matrix fragment (32x16) from row-major K-strided storage (V in attention).
__device__ __forceinline__ v16bf load_b_rowmajor(const __bf16* base, int stride,
                                                 int kbase, int n0, int lane) {
  int n = lane & 15, half = lane >> 4;
  v16bf b;
#pragma unroll
  for (int e = 0; e < 16; ++e)
    b[e] = base[(size_t)(kbase + half * 16 + e) * stride + n0 + n];
  return b;
}

// ---------------------------------------------------------------------------
// fp32 -> bf16 convert (weights)
// ---------------------------------------------------------------------------
__global__ void convert_bf16_kernel(const float* __restrict__ in,
                                    __bf16* __restrict__ out, int n) {
  int i = blockIdx.x * 256 + threadIdx.x;
  if (i < n) out[i] = tobf(in[i]);
}

// ---------------------------------------------------------------------------
// LayerNorm over C=256; one wave32 per row; writes fp32 and bf16 copies.
// ---------------------------------------------------------------------------
__global__ __launch_bounds__(256) void ln_kernel(const float* __restrict__ in,
                                                 const float* __restrict__ w,
                                                 const float* __restrict__ b,
                                                 float* __restrict__ outF,
                                                 __bf16* __restrict__ outB) {
  int row  = blockIdx.x * 8 + (threadIdx.x >> 5);
  int lane = threadIdx.x & 31;
  const float* p = in + (size_t)row * CDIM + lane * 8;
  float v[8];
  float4 f0 = *(const float4*)p;
  float4 f1 = *(const float4*)(p + 4);
  v[0]=f0.x; v[1]=f0.y; v[2]=f0.z; v[3]=f0.w;
  v[4]=f1.x; v[5]=f1.y; v[6]=f1.z; v[7]=f1.w;
  float s = 0.f, sq = 0.f;
#pragma unroll
  for (int i = 0; i < 8; ++i) { s += v[i]; sq += v[i] * v[i]; }
#pragma unroll
  for (int m = 1; m <= 16; m <<= 1) {
    s  += __shfl_xor(s,  m);
    sq += __shfl_xor(sq, m);
  }
  float mean = s * (1.f / CDIM);
  float var  = sq * (1.f / CDIM) - mean * mean;
  float rs   = rsqrtf(var + EPS_LN);
#pragma unroll
  for (int i = 0; i < 8; ++i) {
    int c = lane * 8 + i;
    float o = (v[i] - mean) * rs * w[c] + b[c];
    outF[(size_t)row * CDIM + c] = o;
    outB[(size_t)row * CDIM + c] = tobf(o);
  }
}

// ---------------------------------------------------------------------------
// Tiled WMMA GEMM:  out[M,N] = A[M,K] (bf16) * W[N,K]^T (bf16) + bias [+res]
// BM=128, BN=128, BK=32; 256 threads = 8 waves; wave tile 32x64 (2x4 frags).
// Double-buffered async-to-LDS staging: one s_wait_asynccnt + one barrier
// per K-step.
// ---------------------------------------------------------------------------
enum { EPI_QKV = 0, EPI_RESID = 1, EPI_GELU = 2 };

#define GA_STRIDE 40          // BK=32 + pad 8 (keeps 16B-aligned rows)
#define GA_BUF    (128 * GA_STRIDE)
#define GB_BUF    (128 * GA_STRIDE)

template <int EPI>
__global__ __launch_bounds__(256) void gemm_kernel(
    const __bf16* __restrict__ A, const __bf16* __restrict__ W,
    const float* __restrict__ bias, const float* __restrict__ resid,
    float* __restrict__ outF, __bf16* __restrict__ outB,
    int M, int N, int K) {
  __shared__ __align__(16) __bf16 sA[2 * GA_BUF];  // [buf][m][k]
  __shared__ __align__(16) __bf16 sB[2 * GB_BUF];  // [buf][n][k]

  int tid  = threadIdx.x;
  int wid  = tid >> 5;
  int lane = tid & 31;
  int wm   = (wid & 3) * 32;       // wave M offset in tile
  int wn   = (wid >> 2) * 64;      // wave N offset in tile
  int gm0  = blockIdx.y * 128;
  int gn0  = blockIdx.x * 128;

  // this thread's two A chunks + two B chunks (16B each) per K-step
  int m0 = tid >> 1, ko0 = (tid & 1) * 16;          // covers 128x32 in 2 ld
  const __bf16* gA = A + (size_t)(gm0 + m0) * K + ko0;
  const __bf16* gB = W + (size_t)(gn0 + m0) * K + ko0;

  v8f c[2][4];
#pragma unroll
  for (int i = 0; i < 2; ++i)
#pragma unroll
    for (int j = 0; j < 4; ++j) c[i][j] = (v8f)(0.f);

  const int nk = K >> 5;
  // prologue: stage K-slice 0 into buffer 0
  {
    __bf16* dA = &sA[m0 * GA_STRIDE + ko0];
    __bf16* dB = &sB[m0 * GA_STRIDE + ko0];
    async_b128(dA,     gA);
    async_b128(dA + 8, gA + 8);
    async_b128(dB,     gB);
    async_b128(dB + 8, gB + 8);
  }

  int buf = 0;
  for (int ki = 0; ki < nk; ++ki) {
    wait_async0();
    __syncthreads();
    if (ki + 1 < nk) {
      int k0 = (ki + 1) << 5;
      int nb = buf ^ 1;
      __bf16* dA = &sA[nb * GA_BUF + m0 * GA_STRIDE + ko0];
      __bf16* dB = &sB[nb * GB_BUF + m0 * GA_STRIDE + ko0];
      async_b128(dA,     gA + k0);
      async_b128(dA + 8, gA + k0 + 8);
      async_b128(dB,     gB + k0);
      async_b128(dB + 8, gB + k0 + 8);
    }
    const __bf16* cA = &sA[buf * GA_BUF];
    const __bf16* cB = &sB[buf * GB_BUF];
    v16bf a0 = load_a_frag(cA, GA_STRIDE, wm,      0, lane);
    v16bf a1 = load_a_frag(cA, GA_STRIDE, wm + 16, 0, lane);
#pragma unroll
    for (int j = 0; j < 4; ++j) {
      v16bf bj = load_bt_frag(cB, GA_STRIDE, wn + j * 16, lane);
      c[0][j] = wmma_bf16(a0, bj, c[0][j]);
      c[1][j] = wmma_bf16(a1, bj, c[1][j]);
    }
    buf ^= 1;
  }

  // epilogue: C/D layout -> lane = col (mod 16), VGPR r = row in half
  int n = lane & 15, half = lane >> 4;
#pragma unroll
  for (int i = 0; i < 2; ++i) {
#pragma unroll
    for (int j = 0; j < 4; ++j) {
      int col = gn0 + wn + j * 16 + n;
      float bs = bias[col];
#pragma unroll
      for (int r = 0; r < 8; ++r) {
        int row = gm0 + wm + i * 16 + half * 8 + r;
        float v = c[i][j][r] + bs;
        size_t o = (size_t)row * N + col;
        if (EPI == EPI_QKV) {
          outB[o] = tobf(v);
        } else if (EPI == EPI_RESID) {
          outF[o] = v + resid[o];
        } else {  // EPI_GELU (exact)
          float g = 0.5f * v * (1.f + erff(v * 0.70710678118654752f));
          outB[o] = tobf(g);
        }
      }
    }
  }
}

// ---------------------------------------------------------------------------
// Fused attention per (b, a, h): S = QK^T/sqrt(d), query-row masking,
// softmax in registers, P (bf16) in LDS, Y = P*V.  8 waves, each owns 16
// query rows across all 128 keys.
// ---------------------------------------------------------------------------
__global__ __launch_bounds__(256) void attn_kernel(
    const __bf16* __restrict__ qkv, const unsigned char* __restrict__ mask,
    __bf16* __restrict__ y) {
  __shared__ __align__(16) __bf16 sQ[TSEQ * 40];    // [t][d], pad 8
  __shared__ __align__(16) __bf16 sK[TSEQ * 40];    // [s][d], pad 8
  __shared__ __align__(16) __bf16 sV[TSEQ * 32];    // [s][d]
  __shared__ __align__(16) __bf16 sP[TSEQ * 136];   // [t][s], pad 8
  __shared__ unsigned char sM[TSEQ];

  int h = blockIdx.x, a = blockIdx.y, b = blockIdx.z;
  size_t rowbase = ((size_t)b * 64 + a) * TSEQ;
  int tid  = threadIdx.x;
  int wid  = tid >> 5;
  int lane = tid & 31;

  // stage Q/K/V head tiles via async DMA (16B chunks) + mask
#pragma unroll
  for (int i = 0; i < 2; ++i) {
    int idx = tid + i * 256;
    int tt = idx >> 2, ko = (idx & 3) * 8;
    const __bf16* src = qkv + (rowbase + tt) * C3 + h * DHEAD + ko;
    async_b128(&sQ[tt * 40 + ko], src);
    async_b128(&sK[tt * 40 + ko], src + CDIM);
    async_b128(&sV[tt * 32 + ko], src + 2 * CDIM);
  }
  if (tid < TSEQ) sM[tid] = mask[rowbase + tid];
  wait_async0();
  __syncthreads();

  const int mrow = wid * 16;
  const int n    = lane & 15;
  const int half = lane >> 4;
  const float scale = 0.17677669529663687f;  // 1/sqrt(32)

  // S = Q K^T : one A frag, 8 B frags (keys K-contiguous in sK rows)
  v16bf aq = load_a_frag(sQ, 40, mrow, 0, lane);
  float sv[8][8];  // [key tile j][row r]
#pragma unroll
  for (int j = 0; j < 8; ++j) {
    v16bf bk = load_bt_frag(sK, 40, j * 16, lane);
    v8f s = wmma_bf16(aq, bk, (v8f)(0.f));
#pragma unroll
    for (int r = 0; r < 8; ++r) sv[j][r] = s[r];
  }

  // softmax per query row (row lives in one 16-lane half + 8 frag elems)
#pragma unroll
  for (int r = 0; r < 8; ++r) {
    int m = mrow + half * 8 + r;
    bool mk = sM[m] != 0;   // masked query row -> uniform attention
    float mx = -3.0e38f;
#pragma unroll
    for (int j = 0; j < 8; ++j) {
      float v = mk ? 0.f : sv[j][r] * scale;
      sv[j][r] = v;
      mx = fmaxf(mx, v);
    }
#pragma unroll
    for (int msk = 1; msk <= 8; msk <<= 1) mx = fmaxf(mx, __shfl_xor(mx, msk));
    float sum = 0.f;
#pragma unroll
    for (int j = 0; j < 8; ++j) {
      float e = __expf(sv[j][r] - mx);
      sv[j][r] = e;
      sum += e;
    }
#pragma unroll
    for (int msk = 1; msk <= 8; msk <<= 1) sum += __shfl_xor(sum, msk);
    float inv = 1.f / sum;
#pragma unroll
    for (int j = 0; j < 8; ++j)
      sP[(size_t)m * 136 + j * 16 + n] = tobf(sv[j][r] * inv);
  }
  __syncthreads();

  // Y = P V : K = 128 in 4 steps of 32
  v8f y0 = (v8f)(0.f), y1 = (v8f)(0.f);
#pragma unroll
  for (int ks = 0; ks < 4; ++ks) {
    v16bf ap  = load_a_frag(sP, 136, mrow, ks * 32, lane);
    v16bf bv0 = load_b_rowmajor(sV, 32, ks * 32, 0,  lane);
    v16bf bv1 = load_b_rowmajor(sV, 32, ks * 32, 16, lane);
    y0 = wmma_bf16(ap, bv0, y0);
    y1 = wmma_bf16(ap, bv1, y1);
  }
#pragma unroll
  for (int r = 0; r < 8; ++r) {
    size_t row = rowbase + mrow + half * 8 + r;
    size_t o = row * CDIM + h * DHEAD;
    y[o + n]      = tobf(y0[r]);
    y[o + 16 + n] = tobf(y1[r]);
  }
}

// ---------------------------------------------------------------------------
// Host-side orchestration
// ---------------------------------------------------------------------------
extern "C" void kernel_launch(void* const* d_in, const int* in_sizes, int n_in,
                              void* d_out, int out_size, void* d_ws,
                              size_t ws_size, hipStream_t stream) {
  const float* x      = (const float*)d_in[0];
  const unsigned char* mask = (const unsigned char*)d_in[1];
  const float* ln1_w  = (const float*)d_in[2];
  const float* ln1_b  = (const float*)d_in[3];
  const float* qkv_w  = (const float*)d_in[4];
  const float* qkv_b  = (const float*)d_in[5];
  const float* proj_w = (const float*)d_in[6];
  const float* proj_b = (const float*)d_in[7];
  const float* ln2_w  = (const float*)d_in[8];
  const float* ln2_b  = (const float*)d_in[9];
  const float* fc1_w  = (const float*)d_in[10];
  const float* fc1_b  = (const float*)d_in[11];
  const float* fc2_w  = (const float*)d_in[12];
  const float* fc2_b  = (const float*)d_in[13];
  float* out = (float*)d_out;

  char* ws = (char*)d_ws;
  const size_t szTokC_f  = (size_t)NTOK * CDIM * 4;   // 64 MiB
  const size_t szTokC_b  = (size_t)NTOK * CDIM * 2;   // 32 MiB
  const size_t szTokFF_b = (size_t)NTOK * CFF * 2;    // 128 MiB

  float*  x1f   = (float*)(ws + 0);                       // also x3f later
  __bf16* x1b   = (__bf16*)(ws + szTokC_f);
  __bf16* qkvb  = (__bf16*)(ws + szTokC_f + szTokC_b);    // also h later
  __bf16* yb    = (__bf16*)(ws + szTokC_f + szTokC_b + szTokFF_b); // also x3b
  float*  x2f   = (float*)(ws + szTokC_f + szTokC_b + szTokFF_b + szTokC_b);
  char*   wbase = ws + 2 * szTokC_f + 2 * szTokC_b + szTokFF_b;
  __bf16* qkv_wb  = (__bf16*)(wbase);
  __bf16* proj_wb = (__bf16*)(wbase + (size_t)C3 * CDIM * 2);
  __bf16* fc1_wb  = (__bf16*)(wbase + (size_t)(C3 + CDIM) * CDIM * 2);
  __bf16* fc2_wb  = (__bf16*)(wbase + (size_t)(C3 + CDIM + CFF) * CDIM * 2);
  float*  x3f = x1f;
  __bf16* x3b = yb;
  __bf16* hb  = qkvb;
  (void)in_sizes; (void)n_in; (void)out_size; (void)ws_size;

  // 1) weights -> bf16 (row-major [N][K], K-contiguous for B staging)
  convert_bf16_kernel<<<(C3 * CDIM + 255) / 256, 256, 0, stream>>>(qkv_w, qkv_wb, C3 * CDIM);
  convert_bf16_kernel<<<(CDIM * CDIM + 255) / 256, 256, 0, stream>>>(proj_w, proj_wb, CDIM * CDIM);
  convert_bf16_kernel<<<(CFF * CDIM + 255) / 256, 256, 0, stream>>>(fc1_w, fc1_wb, CFF * CDIM);
  convert_bf16_kernel<<<(CDIM * CFF + 255) / 256, 256, 0, stream>>>(fc2_w, fc2_wb, CDIM * CFF);

  // 2) LN1
  ln_kernel<<<NTOK / 8, 256, 0, stream>>>(x, ln1_w, ln1_b, x1f, x1b);

  // 3) QKV GEMM: [65536,256] x [768,256]^T -> bf16
  gemm_kernel<EPI_QKV><<<dim3(C3 / 128, NTOK / 128), 256, 0, stream>>>(
      x1b, qkv_wb, qkv_b, nullptr, nullptr, qkvb, NTOK, C3, CDIM);

  // 4) attention per (b,a,h)
  attn_kernel<<<dim3(NHEAD, 64, 8), 256, 0, stream>>>(qkvb, mask, yb);

  // 5) proj GEMM + x1 residual -> x2 (fp32)
  gemm_kernel<EPI_RESID><<<dim3(CDIM / 128, NTOK / 128), 256, 0, stream>>>(
      yb, proj_wb, proj_b, x1f, x2f, nullptr, NTOK, CDIM, CDIM);

  // 6) LN2 -> x3 (fp32 + bf16)
  ln_kernel<<<NTOK / 8, 256, 0, stream>>>(x2f, ln2_w, ln2_b, x3f, x3b);

  // 7) fc1 GEMM + GELU -> h (bf16)
  gemm_kernel<EPI_GELU><<<dim3(CFF / 128, NTOK / 128), 256, 0, stream>>>(
      x3b, fc1_wb, fc1_b, nullptr, nullptr, hb, NTOK, CFF, CDIM);

  // 8) fc2 GEMM + x3 residual -> out (fp32)
  gemm_kernel<EPI_RESID><<<dim3(CDIM / 128, NTOK / 128), 256, 0, stream>>>(
      hb, fc2_wb, fc2_b, x3f, out, nullptr, NTOK, CDIM, CFF);
}
